// GNNLayer_32804960207051
// MI455X (gfx1250) — matrix operand order repaired and checked by reading, not compiled, not run on hardware
//
#include <hip/hip_runtime.h>
#include <hip/hip_bf16.h>

#define DEV __device__ __forceinline__

typedef __bf16 bf16_t;
typedef __attribute__((ext_vector_type(16))) __bf16 v16bf;
typedef __attribute__((ext_vector_type(8)))  float  v8f;

static constexpr int Bc  = 8;
static constexpr int Nc  = 4096;
static constexpr int DNc = 64;
static constexpr int Ec  = 1048576;
static constexpr int K1  = 128;   // 2*DN = DN+DM = input K of both MLPs
static constexpr int Hc  = 128;   // hidden width of both MLPs
static constexpr int DMc = 64;    // message width
static constexpr int Fc  = 64;    // output width

static constexpr int NW   = 4;    // waves per block (wave32)
static constexpr int HSTR = 136;  // padded LDS row stride (bf16) to break bank conflicts

// ---------------- WMMA fragment helpers (CDNA5 wave32 layouts) ----------------
// A 16x32 bf16: lane holds row M = lane&15; element j: K = k0 + (lane>>4)*8 + (j<8 ? j : j+8)
DEV v16bf load_fragA_lds(const bf16_t* row, int k0, int lh) {
  v16bf f;
  const bf16_t* p0 = row + k0 + lh * 8;
  const bf16_t* p1 = p0 + 16;
#pragma unroll
  for (int t = 0; t < 8; ++t) { f[t] = p0[t]; f[t + 8] = p1[t]; }
  return f;
}

// A built from two concatenated fp32 rows (src||tgt or node||gathered), with optional scale on 2nd half
DEV v16bf load_fragA_concat(const float* s, const float* t, float tscale, int k0, int lh) {
  v16bf f;
#pragma unroll
  for (int j = 0; j < 16; ++j) {
    int K = k0 + lh * 8 + (j < 8 ? j : j + 8);
    float x = (K < DNc) ? s[K] : t[K - DNc] * tscale;
    f[j] = (bf16_t)x;
  }
  return f;
}

// B 32x16 bf16: lane holds column n = lane&15; element j: K = k0 + (lane>>4)*8 + (j<8 ? j : j+8)
// wt points at the transposed-weight row for column n (contiguous in K, length 128)
DEV v16bf load_fragB(const bf16_t* wt, int k0, int lh) {
  v16bf f;
  const bf16_t* p = wt + k0 + lh * 8;
#pragma unroll
  for (int j = 0; j < 16; ++j) f[j] = p[j < 8 ? j : j + 8];
  return f;
}

DEV v8f wmma_bf16(v16bf a, v16bf b, v8f c) {
  // (neg_a, A, neg_b, B, c_mod, C, reuse_a, reuse_b)
  return __builtin_amdgcn_wmma_f32_16x16x32_bf16(false, a, false, b, (short)0, c, false, false);
}

// ---------------- utility kernels ----------------
__global__ void zero_f(float* __restrict__ p, int n) {
  int i = blockIdx.x * blockDim.x + threadIdx.x;
  if (i < n) p[i] = 0.0f;
}

// Convert + transpose weights to bf16 column-major: Wt[n][k] = W[k][n]
__global__ void prep_weights(const float* __restrict__ Wm1, const float* __restrict__ Wm2,
                             const float* __restrict__ Wu1, const float* __restrict__ Wu2,
                             bf16_t* __restrict__ wm1t, bf16_t* __restrict__ wm2t,
                             bf16_t* __restrict__ wu1t, bf16_t* __restrict__ wu2t) {
  int i = blockIdx.x * blockDim.x + threadIdx.x;
  if (i < 128 * 128) {
    int n = i >> 7, k = i & 127;
    wm1t[i] = (bf16_t)Wm1[k * 128 + n];
    wu1t[i] = (bf16_t)Wu1[k * 128 + n];
  }
  if (i < 64 * 128) {
    int n = i >> 7, k = i & 127;
    wm2t[i] = (bf16_t)Wm2[k * 64 + n];
    wu2t[i] = (bf16_t)Wu2[k * 64 + n];
  }
}

// ---------------- phase 1: per-edge message MLP + weighted segment-sum scatter ----------------
__global__ __launch_bounds__(NW * 32) void msg_kernel(
    const float* __restrict__ node_feats,
    const int*   __restrict__ edge_b,  const int* __restrict__ edge_src,
    const int*   __restrict__ edge_dst, const float* __restrict__ edge_vals,
    const bf16_t* __restrict__ Wm1t, const float* __restrict__ bm1,
    const bf16_t* __restrict__ Wm2t, const float* __restrict__ bm2,
    float* __restrict__ sums, float* __restrict__ counts) {
  __shared__ bf16_t hlds[NW][16][HSTR];
  __shared__ int    seglds[NW][16];

  const int tid  = threadIdx.x;
  const int wave = tid >> 5, lane = tid & 31;
  const int lo = lane & 15, lh = lane >> 4;
  const int e0 = (blockIdx.x * NW + wave) * 16;   // E % 64 == 0, no guards needed

  // indices for the A-row this lane holds (both lane halves hold rows 0..15)
  const int er = e0 + lo;
  const int eb = edge_b[er], es = edge_src[er], ed = edge_dst[er];
  const float* srow = node_feats + ((size_t)eb * Nc + es) * DNc;
  const float* trow = node_feats + ((size_t)eb * Nc + ed) * DNc;
  if (lane < 16) {
    int seg = eb * Nc + es;
    seglds[wave][lo] = seg;
    atomicAdd(&counts[seg], 1.0f);
  }

  // edge weights for the 8 C-rows this lane accumulates (rows lh*8 .. lh*8+7)
  float ev[8];
#pragma unroll
  for (int v = 0; v < 8; ++v) ev[v] = edge_vals[e0 + lh * 8 + v];

  // A = concat(src, tgt) as bf16, 4 K-tiles of 32
  v16bf a[4];
#pragma unroll
  for (int kt = 0; kt < 4; ++kt) a[kt] = load_fragA_concat(srow, trow, 1.0f, kt * 32, lh);

  // layer 1: (16x128)x(128x128) -> relu -> stash bf16 h in LDS (C-layout -> A-layout transpose)
#pragma unroll
  for (int nt = 0; nt < 8; ++nt) {
    v8f c = {};
#pragma unroll
    for (int kt = 0; kt < 4; ++kt) {
      v16bf bfr = load_fragB(Wm1t + (size_t)(nt * 16 + lo) * K1, kt * 32, lh);
      c = wmma_bf16(a[kt], bfr, c);
    }
    float bias = bm1[nt * 16 + lo];
#pragma unroll
    for (int v = 0; v < 8; ++v) {
      float h = c[v] + bias;
      h = h > 0.0f ? h : 0.0f;
      hlds[wave][v + lh * 8][nt * 16 + lo] = (bf16_t)h;
    }
  }
  __syncthreads();

  // layer 2: (16x128)x(128x64) -> messages; weighted atomic scatter into segment sums
  v16bf a2[4];
#pragma unroll
  for (int kt = 0; kt < 4; ++kt) a2[kt] = load_fragA_lds(&hlds[wave][lo][0], kt * 32, lh);

#pragma unroll
  for (int nt = 0; nt < 4; ++nt) {
    v8f c = {};
#pragma unroll
    for (int kt = 0; kt < 4; ++kt) {
      v16bf bfr = load_fragB(Wm2t + (size_t)(nt * 16 + lo) * Hc, kt * 32, lh);
      c = wmma_bf16(a2[kt], bfr, c);
    }
    float bias = bm2[nt * 16 + lo];
#pragma unroll
    for (int v = 0; v < 8; ++v) {
      float m = (c[v] + bias) * ev[v];
      int seg = seglds[wave][v + lh * 8];
      atomicAdd(&sums[(size_t)seg * DMc + nt * 16 + lo], m);
    }
  }
}

// ---------------- phase 2: segment-mean + node-update MLP (sums live in d_out, overwritten) ----
__global__ __launch_bounds__(NW * 32) void upd_kernel(
    const float* __restrict__ node_feats,
    const float* __restrict__ counts,
    const bf16_t* __restrict__ Wu1t, const float* __restrict__ bu1,
    const bf16_t* __restrict__ Wu2t, const float* __restrict__ bu2,
    float* __restrict__ out) {  // in: message sums; out: updated nodes (same rows, same wave)
  __shared__ bf16_t hlds[NW][16][HSTR];

  const int tid  = threadIdx.x;
  const int wave = tid >> 5, lane = tid & 31;
  const int lo = lane & 15, lh = lane >> 4;
  const int r0  = (blockIdx.x * NW + wave) * 16;  // (B*N) % 64 == 0
  const int row = r0 + lo;

  const float invc = 1.0f / fmaxf(counts[row], 1.0f);
  const float* nrow = node_feats + (size_t)row * DNc;
  const float* grow = out + (size_t)row * DMc;    // raw sums -> mean via invc

  v16bf a[4];
#pragma unroll
  for (int kt = 0; kt < 4; ++kt) a[kt] = load_fragA_concat(nrow, grow, invc, kt * 32, lh);

#pragma unroll
  for (int nt = 0; nt < 8; ++nt) {
    v8f c = {};
#pragma unroll
    for (int kt = 0; kt < 4; ++kt) {
      v16bf bfr = load_fragB(Wu1t + (size_t)(nt * 16 + lo) * K1, kt * 32, lh);
      c = wmma_bf16(a[kt], bfr, c);
    }
    float bias = bu1[nt * 16 + lo];
#pragma unroll
    for (int v = 0; v < 8; ++v) {
      float h = c[v] + bias;
      h = h > 0.0f ? h : 0.0f;
      hlds[wave][v + lh * 8][nt * 16 + lo] = (bf16_t)h;
    }
  }
  __syncthreads();

  v16bf a2[4];
#pragma unroll
  for (int kt = 0; kt < 4; ++kt) a2[kt] = load_fragA_lds(&hlds[wave][lo][0], kt * 32, lh);

#pragma unroll
  for (int nt = 0; nt < 4; ++nt) {
    v8f c = {};
#pragma unroll
    for (int kt = 0; kt < 4; ++kt) {
      v16bf bfr = load_fragB(Wu2t + (size_t)(nt * 16 + lo) * Hc, kt * 32, lh);
      c = wmma_bf16(a2[kt], bfr, c);
    }
    float bias = bu2[nt * 16 + lo];
#pragma unroll
    for (int v = 0; v < 8; ++v) {
      out[(size_t)(r0 + lh * 8 + v) * Fc + nt * 16 + lo] = c[v] + bias;
    }
  }
}

// ---------------- host-side launch ----------------
extern "C" void kernel_launch(void* const* d_in, const int* in_sizes, int n_in,
                              void* d_out, int out_size, void* d_ws, size_t ws_size,
                              hipStream_t stream) {
  (void)in_sizes; (void)n_in; (void)out_size; (void)ws_size;

  const float* node_feats = (const float*)d_in[0];
  const int*   edge_b     = (const int*)d_in[1];
  const int*   edge_src   = (const int*)d_in[2];
  const int*   edge_dst   = (const int*)d_in[3];
  const float* edge_vals  = (const float*)d_in[4];
  const float* Wm1 = (const float*)d_in[5];
  const float* bm1 = (const float*)d_in[6];
  const float* Wm2 = (const float*)d_in[7];
  const float* bm2 = (const float*)d_in[8];
  const float* Wu1 = (const float*)d_in[9];
  const float* bu1 = (const float*)d_in[10];
  const float* Wu2 = (const float*)d_in[11];
  const float* bu2 = (const float*)d_in[12];

  float* out = (float*)d_out;                 // (B,N,64): doubles as message-sum accumulator
  char*  ws  = (char*)d_ws;
  float*  counts = (float*)ws;                // B*N floats = 128 KB
  bf16_t* wm1t = (bf16_t*)(ws + 131072);      // 128x128
  bf16_t* wm2t = wm1t + 128 * 128;            // 64x128
  bf16_t* wu1t = wm2t + 64 * 128;             // 128x128
  bf16_t* wu2t = wu1t + 128 * 128;            // 64x128

  zero_f<<<(Bc * Nc * Fc + 255) / 256, 256, 0, stream>>>(out, Bc * Nc * Fc);
  zero_f<<<(Bc * Nc + 255) / 256, 256, 0, stream>>>(counts, Bc * Nc);
  prep_weights<<<(128 * 128 + 255) / 256, 256, 0, stream>>>(Wm1, Wm2, Wu1, Wu2,
                                                            wm1t, wm2t, wu1t, wu2t);

  msg_kernel<<<Ec / (NW * 16), NW * 32, 0, stream>>>(
      node_feats, edge_b, edge_src, edge_dst, edge_vals,
      wm1t, bm1, wm2t, bm2, out, counts);

  upd_kernel<<<(Bc * Nc) / (NW * 16), NW * 32, 0, stream>>>(
      node_feats, counts, wu1t, bu1, wu2t, bu2, out);
}
